// MultiHeadAttention_70506183131547
// MI455X (gfx1250) — matrix operand order repaired and checked by reading, not compiled, not run on hardware
//
#include <hip/hip_runtime.h>
#include <hip/hip_bf16.h>

// ---------------------------------------------------------------------------
// MHA forward for MI455X (gfx1250).
//   - all matmuls via v_wmma_f32_16x16x32_f16 (f32 accumulate)
//   - shared tiles staged in LDS via GLOBAL_LOAD_ASYNC_TO_LDS_B128,
//     double-buffered with s_wait_asynccnt (prefetch distance 2)
// Pipeline:
//   1) cvt fp32->f16 for q,k,v activations
//   2) transpose+cvt Wq,Wk,Wv,Wo -> f16 W^T  (makes WMMA B-fragments contiguous)
//   3) gemm_wmma x3: Q (scaled 1/8, head-major), K (head-major), V ([B,H,Dh,T])
//   4) flash-attention kernel (online softmax, WMMA for QK^T and P*V)
//   5) gemm_wmma: (attn-out f16) @ Wo^T + bo -> fp32 d_out
// ---------------------------------------------------------------------------

typedef _Float16 v16h __attribute__((ext_vector_type(16)));
typedef _Float16 v8h  __attribute__((ext_vector_type(8)));
typedef float    v8f  __attribute__((ext_vector_type(8)));
typedef int      v4i  __attribute__((ext_vector_type(4)));

#define D_MODEL   1024
#define NUM_HEADS 16
#define HEAD_DIM  64
#define BATCH     4
#define SEQ       2048
#define MROWS     (BATCH * SEQ)   // 8192

#define AS1 __attribute__((address_space(1)))
#define AS3 __attribute__((address_space(3)))

// ---- CDNA5 async copy: global -> LDS, 16 bytes per lane ----
__device__ __forceinline__ void async_b128(const void* g, void* l) {
#if __has_builtin(__builtin_amdgcn_global_load_async_to_lds_b128)
  __builtin_amdgcn_global_load_async_to_lds_b128(
      (AS1 v4i*)g, (AS3 v4i*)l, /*offset=*/0, /*cpol=*/0);
#else
  unsigned loff = (unsigned)(unsigned long long)(AS3 void*)l;
  asm volatile("global_load_async_to_lds_b128 %0, %1, off"
               :: "v"(loff), "v"(g) : "memory");
#endif
}

template <int N>
__device__ __forceinline__ void wait_async() {
#if __has_builtin(__builtin_amdgcn_s_wait_asynccnt)
  __builtin_amdgcn_s_wait_asynccnt(N);
#else
  asm volatile("s_wait_asynccnt %0" :: "i"(N) : "memory");
#endif
}

// Load a 16-half A/B fragment as two 16B chunks (offsets in halves, 16B aligned).
__device__ __forceinline__ v16h load_frag(const _Float16* base, int off0, int off1) {
  v8h lo = *(const v8h*)(base + off0);
  v8h hi = *(const v8h*)(base + off1);
  return __builtin_shufflevector(lo, hi, 0,1,2,3,4,5,6,7,8,9,10,11,12,13,14,15);
}

__device__ __forceinline__ v8f wmma16(v16h a, v16h b, v8f c) {
  return __builtin_amdgcn_wmma_f32_16x16x32_f16(
      /*neg_a=*/false, a, /*neg_b=*/false, b,
      /*c_mod=*/(short)0, c, /*reuse_a=*/false, /*reuse_b=*/false);
}

// ---------------------------------------------------------------------------
// fp32 -> f16 elementwise convert
// ---------------------------------------------------------------------------
__global__ __launch_bounds__(256)
void cvt_f32_f16(const float* __restrict__ in, _Float16* __restrict__ out, int n) {
  int i = blockIdx.x * 256 + threadIdx.x;
  if (i < n) out[i] = (_Float16)in[i];
}

// ---------------------------------------------------------------------------
// Wt[n][k] = (f16) W[k][n]   (1024 x 1024)
// ---------------------------------------------------------------------------
__global__ __launch_bounds__(256)
void transpose_w(const float* __restrict__ W, _Float16* __restrict__ Wt) {
  int k = blockIdx.x * 256 + threadIdx.x;  // coalesced write along k
  int n = blockIdx.y;
  Wt[(size_t)n * D_MODEL + k] = (_Float16)W[(size_t)k * D_MODEL + n];
}

// ---------------------------------------------------------------------------
// Y[M=8192, N=1024] = (X[8192,1024] @ W + bias) * scale
//   X row-major f16, Wt = W^T row-major f16.
//   mode 0: f16 out, head-major   [B, H, T, Dh]
//   mode 1: f16 out, head+transp  [B, H, Dh, T]
//   mode 2: f32 out, row-major    [M, N]
// Block = 128 threads (4 waves); wave computes 16(M) x 64(N); K-step 32.
// B tile (64x32 halves, 4KB) async-staged in LDS, double-buffered.
// ---------------------------------------------------------------------------
__global__ __launch_bounds__(128)
void gemm_wmma(const _Float16* __restrict__ X, const _Float16* __restrict__ Wt,
               const float* __restrict__ bias, _Float16* __restrict__ outH,
               float* __restrict__ outF, int mode, float scale)
{
  __shared__ _Float16 bbuf[2][64 * 32];  // two 4KB B tiles

  const int tid  = threadIdx.x;
  const int lane = tid & 31;
  const int wave = tid >> 5;
  const int ln   = lane & 15;
  const int hlf  = lane >> 4;
  const int m0   = blockIdx.x * 64 + wave * 16;
  const int n0   = blockIdx.y * 64;

  // cooperative async copy: 256 chunks x 16B; chunk -> row=ch>>2, col=(ch&3)*8
  const int ch0 = tid, ch1 = tid + 128;
  const int r0 = ch0 >> 2, c0 = (ch0 & 3) * 8;
  const int r1 = ch1 >> 2, c1 = (ch1 & 3) * 8;
  const _Float16* wt0 = Wt + (size_t)(n0 + r0) * D_MODEL + c0;
  const _Float16* wt1 = Wt + (size_t)(n0 + r1) * D_MODEL + c1;

  auto stage = [&](int s) {  // 2 async instructions per wave per stage
    const int kb = s * 32;
    _Float16* dst = &bbuf[s & 1][0];
    async_b128(wt0 + kb, dst + r0 * 32 + c0);
    async_b128(wt1 + kb, dst + r1 * 32 + c1);
  };

  constexpr int NST = D_MODEL / 32;  // 32 k-stages
  stage(0);
  stage(1);

  v8f acc[4] = {};
  const _Float16* arow = X + (size_t)(m0 + ln) * D_MODEL;
  v16h a = load_frag(arow, hlf * 16, hlf * 16 + 8);

  for (int s = 0; s < NST; ++s) {
    if (s + 1 < NST) wait_async<2>();  // current tile landed, next still in flight
    else             wait_async<0>();
    __syncthreads();

    // software-pipeline the per-wave A fragment
    v16h an = a;
    if (s + 1 < NST) {
      const int kb2 = (s + 1) * 32;
      an = load_frag(arow, kb2 + hlf * 16, kb2 + hlf * 16 + 8);
    }

    const _Float16* bb = &bbuf[s & 1][0];
#pragma unroll
    for (int c = 0; c < 4; ++c) {
      v16h b = load_frag(bb + (16 * c + ln) * 32 + hlf * 16, 0, 8);
      acc[c] = wmma16(a, b, acc[c]);
    }
    a = an;

    __syncthreads();                 // everyone done reading this buffer
    if (s + 2 < NST) stage(s + 2);   // safe to overwrite it now
  }

  // Epilogue. C layout: VGPR j -> row (j + 8*half), lane%16 -> column.
#pragma unroll
  for (int c = 0; c < 4; ++c) {
    const int col = n0 + 16 * c + ln;
    const float bv = bias[col];
#pragma unroll
    for (int j = 0; j < 8; ++j) {
      const int   row = m0 + j + 8 * hlf;
      const float val = (acc[c][j] + bv) * scale;
      if (mode == 2) {
        outF[(size_t)row * D_MODEL + col] = val;
      } else {
        const int bb2 = row / SEQ, t = row % SEQ;
        const int hh = col >> 6, d = col & 63;
        size_t idx;
        if (mode == 0)
          idx = (((size_t)bb2 * NUM_HEADS + hh) * SEQ + t) * HEAD_DIM + d;
        else
          idx = (((size_t)bb2 * NUM_HEADS + hh) * HEAD_DIM + d) * SEQ + t;
        outH[idx] = (_Float16)val;
      }
    }
  }
}

// ---------------------------------------------------------------------------
// Flash attention: softmax(Q K^T) V  per (b, h).  Q pre-scaled by 1/sqrt(Dh).
//   Qh, Kh: [B, H, T, 64] f16 ; Vt: [B, H, 64, T] f16 ; out: [B*T, 1024] f16.
// Block = 256 threads (8 waves); wave owns 16 query rows; key blocks of 32.
// K/V tiles async-staged in LDS (shared by all 8 waves), double-buffered.
// ---------------------------------------------------------------------------
__global__ __launch_bounds__(256)
void attn_wmma(const _Float16* __restrict__ Qh, const _Float16* __restrict__ Kh,
               const _Float16* __restrict__ Vt, _Float16* __restrict__ attn)
{
  __shared__ _Float16 kbuf[2][32 * 64];  // 32 keys x 64 d   (4KB each)
  __shared__ _Float16 vbuf[2][64 * 32];  // 64 d   x 32 keys (4KB each)
  __shared__ _Float16 pbuf[8][16 * 40];  // per-wave P tile, stride 40 halves

  const int tid  = threadIdx.x;
  const int lane = tid & 31;
  const int wave = tid >> 5;
  const int ln   = lane & 15;
  const int hlf  = lane >> 4;
  const int h    = blockIdx.y;
  const int b    = blockIdx.z;
  const int q0   = blockIdx.x * 128 + wave * 16;

  const _Float16* Qb = Qh + (((size_t)b * NUM_HEADS + h) * SEQ + q0) * HEAD_DIM;
  const _Float16* Kb = Kh + ((size_t)b * NUM_HEADS + h) * SEQ * HEAD_DIM;
  const _Float16* Vb = Vt + ((size_t)b * NUM_HEADS + h) * HEAD_DIM * SEQ;

  // async copy addressing: K tile 256 chunks (row=tid>>3,col=(tid&7)*8),
  //                        V tile 256 chunks (row=tid>>2,col=(tid&3)*8)
  const int kr = tid >> 3, kc = (tid & 7) * 8;
  const int vr = tid >> 2, vc = (tid & 3) * 8;

  auto stage = [&](int s) {  // 2 async instructions per wave per stage
    const int kb = s * 32;
    async_b128(Kb + (size_t)(kb + kr) * HEAD_DIM + kc, &kbuf[s & 1][kr * 64 + kc]);
    async_b128(Vb + (size_t)vr * SEQ + kb + vc,        &vbuf[s & 1][vr * 32 + vc]);
  };

  // Q A-fragments for d-chunks [0..31] and [32..63] (reused all key blocks)
  const _Float16* qrow = Qb + (size_t)ln * HEAD_DIM;
  const v16h qa0 = load_frag(qrow, 0  + hlf * 16, 0  + hlf * 16 + 8);
  const v16h qa1 = load_frag(qrow, 32 + hlf * 16, 32 + hlf * 16 + 8);

  v8f o[4] = {};
  float mrow[8], lrow[8];
#pragma unroll
  for (int j = 0; j < 8; ++j) { mrow[j] = -__builtin_inff(); lrow[j] = 0.f; }

  _Float16* pl = &pbuf[wave][0];

  constexpr int NST = SEQ / 32;  // 64 key stages
  stage(0);
  stage(1);

  for (int s = 0; s < NST; ++s) {
    if (s + 1 < NST) wait_async<2>();
    else             wait_async<0>();
    __syncthreads();

    const _Float16* kB = &kbuf[s & 1][0];
    const _Float16* vB = &vbuf[s & 1][0];

    // ---- S = Q K^T for two 16-key sub-tiles (4 WMMAs, B-frags from LDS) ----
    v8f s0 = {}, s1 = {};
    {
      const _Float16* k0 = kB + (size_t)ln * HEAD_DIM + hlf * 16;
      s0 = wmma16(qa0, load_frag(k0, 0, 8),   s0);
      s0 = wmma16(qa1, load_frag(k0, 32, 40), s0);
      const _Float16* k1 = kB + (size_t)(16 + ln) * HEAD_DIM + hlf * 16;
      s1 = wmma16(qa0, load_frag(k1, 0, 8),   s1);
      s1 = wmma16(qa1, load_frag(k1, 32, 40), s1);
    }

    // ---- online softmax (row reductions across the 16-lane half) ----
    float p0[8], p1[8], scl[8];
#pragma unroll
    for (int j = 0; j < 8; ++j) {
      float rm = fmaxf(s0[j], s1[j]);
      rm = fmaxf(rm, __shfl_xor(rm, 1, 32));
      rm = fmaxf(rm, __shfl_xor(rm, 2, 32));
      rm = fmaxf(rm, __shfl_xor(rm, 4, 32));
      rm = fmaxf(rm, __shfl_xor(rm, 8, 32));
      const float mn = fmaxf(mrow[j], rm);
      scl[j] = __expf(mrow[j] - mn);
      mrow[j] = mn;
      p0[j] = __expf(s0[j] - mn);
      p1[j] = __expf(s1[j] - mn);
      float rs = p0[j] + p1[j];
      rs += __shfl_xor(rs, 1, 32);
      rs += __shfl_xor(rs, 2, 32);
      rs += __shfl_xor(rs, 4, 32);
      rs += __shfl_xor(rs, 8, 32);
      lrow[j] = lrow[j] * scl[j] + rs;
    }

    // rescale O, spill P (C-layout) to LDS as f16 (wave-private buffer)
#pragma unroll
    for (int j = 0; j < 8; ++j) {
      o[0][j] *= scl[j]; o[1][j] *= scl[j]; o[2][j] *= scl[j]; o[3][j] *= scl[j];
      const int row = j + 8 * hlf;
      pl[row * 40 + ln]      = (_Float16)p0[j];
      pl[row * 40 + 16 + ln] = (_Float16)p1[j];
    }

    // reload P as A-fragment (LDS ops are in-order within a wave)
    const _Float16* pr = pl + (size_t)ln * 40 + hlf * 16;
    v8h plo = *(const v8h*)(pr);
    v8h phi = *(const v8h*)(pr + 8);
    v16h pa = __builtin_shufflevector(plo, phi,
                                      0,1,2,3,4,5,6,7,8,9,10,11,12,13,14,15);

    // ---- O += P @ V  (4 WMMAs, B-frags from LDS V^T tile) ----
#pragma unroll
    for (int c = 0; c < 4; ++c) {
      o[c] = wmma16(pa, load_frag(vB + (16 * c + ln) * 32 + hlf * 16, 0, 8), o[c]);
    }

    __syncthreads();                 // all waves done with this K/V buffer
    if (s + 2 < NST) stage(s + 2);   // overwrite it for stage s+2
  }

  // ---- normalize and write attention output (f16, [B*T, D_MODEL]) ----
  float rl[8];
#pragma unroll
  for (int j = 0; j < 8; ++j) rl[j] = 1.0f / lrow[j];
  _Float16* ob = attn + ((size_t)b * SEQ + q0) * D_MODEL + h * HEAD_DIM;
#pragma unroll
  for (int c = 0; c < 4; ++c) {
#pragma unroll
    for (int j = 0; j < 8; ++j) {
      const int row = j + 8 * hlf;
      ob[(size_t)row * D_MODEL + 16 * c + ln] = (_Float16)(o[c][j] * rl[j]);
    }
  }
}

// ---------------------------------------------------------------------------
extern "C" void kernel_launch(void* const* d_in, const int* in_sizes, int n_in,
                              void* d_out, int out_size, void* d_ws, size_t ws_size,
                              hipStream_t stream) {
  (void)in_sizes; (void)n_in; (void)out_size; (void)ws_size;
  const float* q  = (const float*)d_in[0];
  const float* k  = (const float*)d_in[1];
  const float* v  = (const float*)d_in[2];
  const float* Wq = (const float*)d_in[3];
  const float* bq = (const float*)d_in[4];
  const float* Wk = (const float*)d_in[5];
  const float* bk = (const float*)d_in[6];
  const float* Wv = (const float*)d_in[7];
  const float* bv = (const float*)d_in[8];
  const float* Wo = (const float*)d_in[9];
  const float* bo = (const float*)d_in[10];
  float* out = (float*)d_out;

  const size_t SZ = (size_t)MROWS * D_MODEL;      // 8,388,608 halves
  const size_t WS = (size_t)D_MODEL * D_MODEL;    // 1,048,576 halves
  _Float16* ws  = (_Float16*)d_ws;
  _Float16* Xq  = ws;            // activations in f16
  _Float16* Xk  = Xq + SZ;
  _Float16* Xv  = Xk + SZ;
  _Float16* WqT = Xv + SZ;       // transposed weights in f16
  _Float16* WkT = WqT + WS;
  _Float16* WvT = WkT + WS;
  _Float16* WoT = WvT + WS;
  _Float16* Qh  = WoT + WS;      // [B,H,T,Dh], pre-scaled by 1/8
  _Float16* Kh  = Qh + SZ;       // [B,H,T,Dh]
  _Float16* Vt  = Kh + SZ;       // [B,H,Dh,T]
  _Float16* At  = Vt + SZ;       // attention output, [B*T, D_MODEL]

  // 1) activation converts
  cvt_f32_f16<<<(int)(SZ / 256), 256, 0, stream>>>(q, Xq, (int)SZ);
  cvt_f32_f16<<<(int)(SZ / 256), 256, 0, stream>>>(k, Xk, (int)SZ);
  cvt_f32_f16<<<(int)(SZ / 256), 256, 0, stream>>>(v, Xv, (int)SZ);

  // 2) weight transposes
  dim3 tg(D_MODEL / 256, D_MODEL);
  transpose_w<<<tg, 256, 0, stream>>>(Wq, WqT);
  transpose_w<<<tg, 256, 0, stream>>>(Wk, WkT);
  transpose_w<<<tg, 256, 0, stream>>>(Wv, WvT);
  transpose_w<<<tg, 256, 0, stream>>>(Wo, WoT);

  // 3) projections (softmax scale folded into Q)
  dim3 gg(MROWS / 64, D_MODEL / 64);
  gemm_wmma<<<gg, 128, 0, stream>>>(Xq, WqT, bq, Qh, nullptr, 0, 0.125f);
  gemm_wmma<<<gg, 128, 0, stream>>>(Xk, WkT, bk, Kh, nullptr, 0, 1.0f);
  gemm_wmma<<<gg, 128, 0, stream>>>(Xv, WvT, bv, Vt, nullptr, 1, 1.0f);

  // 4) attention
  dim3 ag(SEQ / 128, NUM_HEADS, BATCH);
  attn_wmma<<<ag, 256, 0, stream>>>(Qh, Kh, Vt, At);

  // 5) output projection -> fp32 d_out
  gemm_wmma<<<gg, 128, 0, stream>>>(At, WoT, bo, nullptr, out, 2, 1.0f);
}
